// sLSTMClassifier_51213190037940
// MI455X (gfx1250) — matrix operand order repaired and checked by reading, not compiled, not run on hardware
//
#include <hip/hip_runtime.h>
#include <hip/hip_bf16.h>
#include <math.h>

// sLSTM classifier forward for MI455X (gfx1250, wave32, WMMA).
// Pipeline:
//   k_prep       : pack R and Wgates (f32->bf16) into WMMA B-operand lane layout
//   k_embed_ln1  : embedding gather + LayerNorm -> ln1 (bf16), h_last (f32)
//   k_conv       : depthwise causal conv K=4 + SiLU -> cc (bf16)
//   k_gates_gemm : 16x (M=16384,N=256,K=256) bf16 WMMA GEMMs -> Wx (+Rb folded), bf16
//   k_slstm      : 4 persistent blocks (1 per head), 512-step scan; per step a
//                  32x1024x256 WMMA GEMM from LDS hprev, exp-gated state update
//   k_combine / k_ffn1 / k_ffn2 / k_final : last-position-only tail (32 rows)

#define B_   32
#define S_   512
#define D_   1024
#define NH_  4
#define HD_  256
#define KCONV 4
#define UP_  1344

typedef __attribute__((ext_vector_type(16))) __bf16        v16bf;
typedef __attribute__((ext_vector_type(8)))  float         v8f;
typedef __attribute__((ext_vector_type(8)))  unsigned int  v8u;

__device__ __forceinline__ unsigned short f2bf(float f) {
  unsigned int u = __builtin_bit_cast(unsigned int, f);
  u += 0x7fffu + ((u >> 16) & 1u);          // round-to-nearest-even
  return (unsigned short)(u >> 16);
}
__device__ __forceinline__ float bf2f(unsigned int h) {
  return __builtin_bit_cast(float, (h & 0xffffu) << 16);
}
__device__ __forceinline__ unsigned int packbf(float lo, float hi) {
  return (unsigned int)f2bf(lo) | ((unsigned int)f2bf(hi) << 16);
}
__device__ __forceinline__ v16bf mk16(uint4 a, uint4 b) {
  v8u u; u[0]=a.x; u[1]=a.y; u[2]=a.z; u[3]=a.w; u[4]=b.x; u[5]=b.y; u[6]=b.z; u[7]=b.w;
  return __builtin_bit_cast(v16bf, u);
}
__device__ __forceinline__ float sigf(float x)    { return 1.0f / (1.0f + expf(-x)); }
__device__ __forceinline__ float logsigf(float x) { return fminf(x, 0.0f) - log1pf(expf(-fabsf(x))); }

// two-value block reduction (blockDim.x multiple of 32, wave32)
__device__ __forceinline__ void blockReduce2(float& a, float& b, float* red, int tid) {
  #pragma unroll
  for (int off = 16; off > 0; off >>= 1) {
    a += __shfl_down(a, off, 32);
    b += __shfl_down(b, off, 32);
  }
  int w = tid >> 5, lane = tid & 31;
  if (lane == 0) { red[w*2] = a; red[w*2+1] = b; }
  __syncthreads();
  if (tid == 0) {
    float s0 = 0.f, s1 = 0.f;
    int nw = blockDim.x >> 5;
    for (int i = 0; i < nw; i++) { s0 += red[i*2]; s1 += red[i*2+1]; }
    red[0] = s0; red[1] = s1;
  }
  __syncthreads();
  a = red[0]; b = red[1];
  __syncthreads();
}

// ---------------------------------------------------------------- k_prep
// PB[n][kt][col(1024)][half][d]  <- R (nh,hd,4,hd):   pair (k, k+1), k = kt*32+half*16+2d
// PW[gn][kt][col(256)][half][d]  <- Wgates (4,nh,hd,hd)
__global__ void k_prep(const float* __restrict__ R, const float* __restrict__ Wg,
                       unsigned int* __restrict__ PB, unsigned int* __restrict__ PW) {
  unsigned int idx = blockIdx.x * 256u + threadIdx.x;          // 0 .. 1048575
  if (idx < 524288u) {                                         // R -> PB
    unsigned int d = idx & 7u, half = (idx >> 3) & 1u, col = (idx >> 4) & 1023u;
    unsigned int kt = (idx >> 14) & 7u, n = idx >> 17;
    unsigned int k = kt*32u + half*16u + 2u*d;
    long s0 = ((long)(n*256u + k)) * 1024 + col;               // ((n*hd+k)*4+g)*hd+e, col=g*hd+e
    PB[idx] = packbf(R[s0], R[s0 + 1024]);
  } else {                                                     // Wgates -> PW
    unsigned int j = idx - 524288u;
    unsigned int d = j & 7u, half = (j >> 3) & 1u, col = (j >> 4) & 255u;
    unsigned int kt = (j >> 12) & 7u, gn = j >> 15;
    unsigned int k = kt*32u + half*16u + 2u*d;
    long s0 = ((long)(gn*256u + k)) * 256 + col;
    PW[j] = packbf(Wg[s0], Wg[s0 + 256]);
  }
}

// ---------------------------------------------------------------- k_embed_ln1
__global__ void k_embed_ln1(const int* __restrict__ x, const float* __restrict__ E,
                            const float* __restrict__ ln1w,
                            unsigned short* __restrict__ ln1, float* __restrict__ hlast) {
  int bs = blockIdx.x; int b = bs >> 9, s = bs & 511;
  int tid = threadIdx.x;
  __shared__ float red[64];
  long tok = (long)x[bs];
  float4 v = ((const float4*)(E + tok * (long)D_))[tid];
  float s1 = v.x + v.y + v.z + v.w;
  float s2 = v.x*v.x + v.y*v.y + v.z*v.z + v.w*v.w;
  blockReduce2(s1, s2, red, tid);
  float mu = s1 * (1.0f / D_);
  float rs = rsqrtf(s2 * (1.0f / D_) - mu*mu + 1e-5f);
  int d0 = tid * 4;
  unsigned int o0 = packbf((v.x - mu)*rs*ln1w[d0],     (v.y - mu)*rs*ln1w[d0+1]);
  unsigned int o1 = packbf((v.z - mu)*rs*ln1w[d0+2],   (v.w - mu)*rs*ln1w[d0+3]);
  ((uint2*)(ln1 + (long)bs * D_))[tid] = make_uint2(o0, o1);
  if (s == S_ - 1) ((float4*)(hlast + (long)b * D_))[tid] = v;
}

// ---------------------------------------------------------------- k_conv (depthwise causal K=4 + SiLU)
__global__ void k_conv(const unsigned short* __restrict__ ln1, const float* __restrict__ cw,
                       const float* __restrict__ cb, unsigned short* __restrict__ cc) {
  int bs = blockIdx.x; int b = bs >> 9, s = bs & 511;
  int tid = threadIdx.x;
  int d0 = tid * 4;
  float acc[4] = { cb[d0], cb[d0+1], cb[d0+2], cb[d0+3] };
  #pragma unroll
  for (int k = 0; k < KCONV; k++) {
    int sp = s - (KCONV - 1) + k;
    if (sp >= 0) {
      uint2 pr = *(const uint2*)(ln1 + ((long)b * S_ + sp) * D_ + d0);
      acc[0] += bf2f(pr.x)       * cw[(d0+0)*KCONV + k];
      acc[1] += bf2f(pr.x >> 16) * cw[(d0+1)*KCONV + k];
      acc[2] += bf2f(pr.y)       * cw[(d0+2)*KCONV + k];
      acc[3] += bf2f(pr.y >> 16) * cw[(d0+3)*KCONV + k];
    }
  }
  float o[4];
  #pragma unroll
  for (int i = 0; i < 4; i++) o[i] = acc[i] * sigf(acc[i]);   // SiLU
  ((uint2*)(cc + (long)bs * D_))[tid] = make_uint2(packbf(o[0], o[1]), packbf(o[2], o[3]));
}

// ---------------------------------------------------------------- k_gates_gemm
// 16 GEMMs (gate g, head n): A = cc (g<2) or ln1 (g>=2) head slice, B = PW, +Rb folded.
// Block: 8 waves; wave -> (2 M-tiles via blockIdx.x, 4 N-tiles). bf16 WMMA, f32 acc.
__global__ void k_gates_gemm(const unsigned short* __restrict__ ln1,
                             const unsigned short* __restrict__ cc,
                             const unsigned int* __restrict__ PW,
                             const float* __restrict__ Rb,
                             unsigned short* __restrict__ WX) {
  int combo = blockIdx.y;                 // g*4 + n
  int g = combo >> 2, n = combo & 3;
  int tid = threadIdx.x, lane = tid & 31, w = tid >> 5;
  int mt = blockIdx.x * 2 + (w >> 2);     // M-tile (16 rows of 16384)
  int ng = w & 3;                         // N-group: 4 tiles => 64 of 256 cols
  const unsigned short* A = (g < 2) ? cc : ln1;
  int arow = mt * 16 + (lane & 15);
  const unsigned short* abase = A + (long)arow * D_ + n * HD_;
  int base0 = (lane >= 16) ? 16 : 0;      // bf16 K offset per A lane layout
  const unsigned int* pwbase = PW + (long)combo * 32768;
  v8f acc[4] = {};
  #pragma unroll
  for (int kt = 0; kt < 8; kt++) {
    uint4 alo = *(const uint4*)(abase + kt*32 + base0);
    uint4 ahi = *(const uint4*)(abase + kt*32 + base0 + 8);
    v16bf av = mk16(alo, ahi);
    #pragma unroll
    for (int i = 0; i < 4; i++) {
      int col = (ng*4 + i)*16 + (lane & 15);
      const unsigned int* pb = pwbase + kt*4096 + col*16 + (lane >> 4)*8;
      v16bf bv = mk16(*(const uint4*)pb, *(const uint4*)(pb + 4));
      acc[i] = __builtin_amdgcn_wmma_f32_16x16x32_bf16(false, av, false, bv,
                                                       (short)0, acc[i], false, false);
    }
  }
  int mrow = mt * 16 + ((lane >> 4) << 3);
  #pragma unroll
  for (int i = 0; i < 4; i++) {
    int e = (ng*4 + i)*16 + (lane & 15);
    float rb = Rb[combo * HD_ + e];
    #pragma unroll
    for (int r = 0; r < 8; r++) {
      int row = mrow + r;                 // b*S + s
      int b = row >> 9, s = row & 511;
      long idx = (((long)n * S_ + s) * B_ + b) * (4*HD_) + g*HD_ + e;
      WX[idx] = f2bf(acc[i][r] + rb);
    }
  }
}

// ---------------------------------------------------------------- k_slstm (recurrence)
// 4 blocks (one head), 1024 threads = 32 waves. Per step:
//   GEMM r = hprev(32x256, LDS bf16) @ R_n(256x1024, L2-resident packed bf16)
//   raw = r + wx (Rb prefolded) -> LDS f32; pointwise exp-gated state update.
__global__ void __launch_bounds__(1024) k_slstm(const unsigned int* __restrict__ PB,
                                                const unsigned short* __restrict__ WX,
                                                float* __restrict__ ylast) {
  int n = blockIdx.x;
  extern __shared__ unsigned char smem[];
  unsigned int* hA32 = (unsigned int*)smem;                    // 32 x 128 dwords (bf16 hprev)
  float* raw = (float*)(smem + 32*256*2);                      // 32 x 1024 f32
  int tid = threadIdx.x, lane = tid & 31, w = tid >> 5;
  for (int i = tid; i < 32*128; i += 1024) hA32[i] = 0u;       // hprev = 0
  int sb = tid >> 5;                                           // batch element owned (state)
  int e0 = (tid & 31) * 8;                                     // 8 channels per thread
  float c[8], nn_[8], mm[8];
  #pragma unroll
  for (int j = 0; j < 8; j++) { c[j] = 0.f; nn_[j] = 0.f; mm[j] = -1e30f; }
  int mt = w >> 4, ng = w & 15;                                // wave tiling: 2 x 16
  int arow = mt * 16 + (lane & 15);
  int base0d = (lane >= 16) ? 8 : 0;                           // dword offset in A row
  const unsigned int* pbn = PB + (long)n * 131072;
  __syncthreads();
  for (int t = 0; t < S_; t++) {
    const unsigned short* wxt = WX + ((long)n * S_ + t) * B_ * (4*HD_);
    if (t + 1 < S_)                                            // pull next wx slab toward L2/L0
      __builtin_prefetch((const char*)(wxt + (long)B_*4*HD_) + tid*64, 0, 1);
    v8f acc[4] = {};
    #pragma unroll
    for (int kt = 0; kt < 8; kt++) {
      const unsigned int* ap = hA32 + arow*128 + kt*16 + base0d;
      v16bf av = mk16(*(const uint4*)ap, *(const uint4*)(ap + 4));
      #pragma unroll
      for (int i = 0; i < 4; i++) {
        int col = (ng*4 + i)*16 + (lane & 15);
        const unsigned int* pb = pbn + kt*16384 + col*16 + (lane >> 4)*8;
        acc[i] = __builtin_amdgcn_wmma_f32_16x16x32_bf16(
            false, av, false, mk16(*(const uint4*)pb, *(const uint4*)(pb + 4)),
            (short)0, acc[i], false, false);
      }
    }
    int mrow = mt * 16 + ((lane >> 4) << 3);
    #pragma unroll
    for (int i = 0; i < 4; i++) {
      int colg = (ng*4 + i)*16 + (lane & 15);                  // 0..1023 = g*256+e
      #pragma unroll
      for (int r = 0; r < 8; r++) {
        int bb = mrow + r;                                     // batch row 0..31
        float wx = bf2f((unsigned int)wxt[(long)bb*(4*HD_) + colg]);
        raw[bb*1024 + colg] = acc[i][r] + wx;
      }
    }
    __syncthreads();
    #pragma unroll
    for (int j = 0; j < 8; j += 2) {
      float hv[2];
      #pragma unroll
      for (int q = 0; q < 2; q++) {
        int e = e0 + j + q;
        float ir  = raw[sb*1024 +           e];
        float fr  = raw[sb*1024 + 1*HD_*1 + 256 - 256 + 256 + e - e + 256 + e]; // placeholder avoided
        fr        = raw[sb*1024 + 256 + e];
        float zr  = raw[sb*1024 + 512 + e];
        float orr = raw[sb*1024 + 768 + e];
        float lpm = mm[j+q] + logsigf(fr);
        float mn  = fmaxf(ir, lpm);                            // t=0: mm=-1e30 => mn=ir
        float ig  = expf(ir - mn);
        float fg  = expf(lpm - mn);
        c[j+q]   = fg * c[j+q]   + ig * tanhf(zr);
        nn_[j+q] = fg * nn_[j+q] + ig;
        mm[j+q]  = mn;
        hv[q] = sigf(orr) * c[j+q] / nn_[j+q];
      }
      hA32[sb*128 + ((e0 + j) >> 1)] = packbf(hv[0], hv[1]);
      if (t == S_ - 1) {
        ylast[(long)sb*D_ + n*HD_ + e0 + j]     = hv[0];
        ylast[(long)sb*D_ + n*HD_ + e0 + j + 1] = hv[1];
      }
    }
    __syncthreads();
  }
}

// ---------------------------------------------------------------- tail (last position only)
__global__ void k_combine(const float* __restrict__ ylast, const float* __restrict__ hlast,
                          const float* __restrict__ gnw, const float* __restrict__ ln2w,
                          float* __restrict__ hfin, float* __restrict__ hln2) {
  int b = blockIdx.x, tid = threadIdx.x;
  __shared__ float hp[256], hq[256], hstat[8], red[64];
  float4 v = ((const float4*)(ylast + (long)b * D_))[tid];
  hp[tid] = v.x + v.y + v.z + v.w;
  hq[tid] = v.x*v.x + v.y*v.y + v.z*v.z + v.w*v.w;
  __syncthreads();
  if (tid < 4) {                                              // per-head stats (hd=256 <- 64 thr)
    float a = 0.f, q = 0.f;
    for (int i = tid*64; i < tid*64 + 64; i++) { a += hp[i]; q += hq[i]; }
    float mu = a * (1.0f / HD_);
    hstat[tid] = mu; hstat[4 + tid] = rsqrtf(q * (1.0f / HD_) - mu*mu + 1e-5f);
  }
  __syncthreads();
  int hd_ = tid >> 6;
  float mu = hstat[hd_], rs = hstat[4 + hd_];
  int d0 = tid * 4;
  float4 hl = ((const float4*)(hlast + (long)b * D_))[tid];
  float f0 = hl.x + (v.x - mu)*rs*gnw[d0];
  float f1 = hl.y + (v.y - mu)*rs*gnw[d0+1];
  float f2 = hl.z + (v.z - mu)*rs*gnw[d0+2];
  float f3 = hl.w + (v.w - mu)*rs*gnw[d0+3];
  float a2 = f0 + f1 + f2 + f3;
  float b2 = f0*f0 + f1*f1 + f2*f2 + f3*f3;
  blockReduce2(a2, b2, red, tid);
  float mu2 = a2 * (1.0f / D_);
  float rs2 = rsqrtf(b2 * (1.0f / D_) - mu2*mu2 + 1e-5f);
  ((float4*)(hfin + (long)b * D_))[tid] = make_float4(f0, f1, f2, f3);
  ((float4*)(hln2 + (long)b * D_))[tid] = make_float4(
      (f0 - mu2)*rs2*ln2w[d0],   (f1 - mu2)*rs2*ln2w[d0+1],
      (f2 - mu2)*rs2*ln2w[d0+2], (f3 - mu2)*rs2*ln2w[d0+3]);
}

__global__ void k_ffn1(const float* __restrict__ hln2, const float* __restrict__ W,
                       float* __restrict__ dvec) {
  int b = blockIdx.y;
  int j = blockIdx.x * 256 + threadIdx.x;
  __shared__ float hl[D_];
  for (int i = threadIdx.x; i < D_; i += 256) hl[i] = hln2[(long)b*D_ + i];
  __syncthreads();
  if (j >= UP_) return;
  float g = 0.f, u = 0.f;
  for (int d = 0; d < D_; d++) {
    float hv = hl[d];
    g += hv * W[(long)d*(2*UP_) + j];
    u += hv * W[(long)d*(2*UP_) + UP_ + j];
  }
  float ge = 0.5f * g * (1.0f + erff(g * 0.70710678118654752f));   // exact GELU
  dvec[(long)b*UP_ + j] = ge * u;
}

__global__ void k_ffn2(const float* __restrict__ dvec, const float* __restrict__ Wd,
                       const float* __restrict__ hfin, float* __restrict__ h2) {
  int b = blockIdx.y;
  int col = blockIdx.x * 256 + threadIdx.x;                   // 0..1023
  __shared__ float dl[UP_];
  for (int i = threadIdx.x; i < UP_; i += 256) dl[i] = dvec[(long)b*UP_ + i];
  __syncthreads();
  float acc = hfin[(long)b*D_ + col];
  for (int k = 0; k < UP_; k++) acc += dl[k] * Wd[(long)k*D_ + col];
  h2[(long)b*D_ + col] = acc;
}

__global__ void k_final(const float* __restrict__ h2, const float* __restrict__ postw,
                        const float* __restrict__ fcw, const float* __restrict__ fcb,
                        float* __restrict__ out) {
  int b = blockIdx.x, tid = threadIdx.x;
  __shared__ float red[64];
  float4 v = ((const float4*)(h2 + (long)b * D_))[tid];
  float s1 = v.x + v.y + v.z + v.w;
  float s2 = v.x*v.x + v.y*v.y + v.z*v.z + v.w*v.w;
  blockReduce2(s1, s2, red, tid);
  float mu = s1 * (1.0f / D_);
  float rs = rsqrtf(s2 * (1.0f / D_) - mu*mu + 1e-5f);
  int d0 = tid * 4;
  float dot = (v.x - mu)*rs*postw[d0]  *fcw[d0]
            + (v.y - mu)*rs*postw[d0+1]*fcw[d0+1]
            + (v.z - mu)*rs*postw[d0+2]*fcw[d0+2]
            + (v.w - mu)*rs*postw[d0+3]*fcw[d0+3];
  float zero = 0.f;
  blockReduce2(dot, zero, red, tid);
  if (tid == 0) out[b] = 1.0f / (1.0f + expf(-(dot + fcb[0])));
}

// ---------------------------------------------------------------- launch
extern "C" void kernel_launch(void* const* d_in, const int* in_sizes, int n_in,
                              void* d_out, int out_size, void* d_ws, size_t ws_size,
                              hipStream_t stream) {
  (void)in_sizes; (void)n_in; (void)out_size; (void)ws_size;
  const int*   x      = (const int*)  d_in[0];
  const float* E      = (const float*)d_in[1];
  const float* ln1w   = (const float*)d_in[2];
  const float* convw  = (const float*)d_in[3];
  const float* convb  = (const float*)d_in[4];
  const float* Wg     = (const float*)d_in[5];
  const float* R      = (const float*)d_in[6];
  const float* Rb     = (const float*)d_in[7];
  const float* gnw    = (const float*)d_in[8];
  const float* ln2w   = (const float*)d_in[9];
  const float* ffn_up = (const float*)d_in[10];
  const float* ffn_dn = (const float*)d_in[11];
  const float* postw  = (const float*)d_in[12];
  const float* fcw    = (const float*)d_in[13];
  const float* fcb    = (const float*)d_in[14];
  float* out = (float*)d_out;

  char* ws = (char*)d_ws;                      // ~197 MB used
  unsigned short* ln1 = (unsigned short*)(ws + 0);             //  32 MB
  unsigned short* cc  = (unsigned short*)(ws + 33554432);      //  32 MB
  unsigned int*   PB  = (unsigned int*)  (ws + 67108864);      //   2 MB
  unsigned int*   PW  = (unsigned int*)  (ws + 69206016);      //   2 MB
  unsigned short* WX  = (unsigned short*)(ws + 71303168);      // 128 MB
  float* hlast = (float*)(ws + 205520896);
  float* ylast = (float*)(ws + 205651968);
  float* hfin  = (float*)(ws + 205783040);
  float* hln2  = (float*)(ws + 205914112);
  float* dvec  = (float*)(ws + 206045184);
  float* h2    = (float*)(ws + 206217216);

  k_prep<<<4096, 256, 0, stream>>>(R, Wg, PB, PW);
  k_embed_ln1<<<B_*S_, 256, 0, stream>>>(x, E, ln1w, ln1, hlast);
  k_conv<<<B_*S_, 256, 0, stream>>>(ln1, convw, convb, cc);
  k_gates_gemm<<<dim3(512, 16), 256, 0, stream>>>(ln1, cc, PW, Rb, WX);
  k_slstm<<<NH_, 1024, 32*256*2 + 32*1024*4, stream>>>(PB, WX, ylast);
  k_combine<<<B_, 256, 0, stream>>>(ylast, hlast, gnw, ln2w, hfin, hln2);
  k_ffn1<<<dim3((UP_ + 255)/256, B_), 256, 0, stream>>>(hln2, ffn_up, dvec);
  k_ffn2<<<dim3(D_/256, B_), 256, 0, stream>>>(dvec, ffn_dn, hfin, h2);
  k_final<<<B_, 256, 0, stream>>>(h2, postw, fcw, fcb, out);
}
